// SerriformNet_88716844466866
// MI455X (gfx1250) — matrix operand order repaired and checked by reading, not compiled, not run on hardware
//
#include <hip/hip_runtime.h>
#include <hip/hip_bf16.h>
#include <math.h>

// ---- problem constants (match reference) ----
#define B_    2
#define L_    2048
#define D_    512
#define V_    50257
#define NL_   12
#define K_    5
#define A_    128
#define H_    128
#define HALF_ 256
#define ML    (B_ * L_)   // 4096 rows

// ---- CDNA5 feature probes (compile-time, target-gated) ----
#if defined(__AMDGCN__) && __has_builtin(__builtin_amdgcn_global_load_async_to_lds_b128)
#define HAVE_ASYNC 1
#else
#define HAVE_ASYNC 0
#endif

// async builtin takes pointers to 16B int vectors (per clang diagnostic)
typedef int v4i __attribute__((vector_size(16)));
#define AS1V(p) ((__attribute__((address_space(1))) v4i*)(p))
#define AS3V(p) ((__attribute__((address_space(3))) v4i*)(p))

// ---- WMMA vector types (gfx1250, wave32) ----
typedef __attribute__((ext_vector_type(16))) __bf16        v16bf;
typedef __attribute__((ext_vector_type(2)))  __bf16        v2bf;
typedef __attribute__((ext_vector_type(8)))  float         v8f;
typedef __attribute__((ext_vector_type(8)))  unsigned int  v8u;

__device__ __forceinline__ unsigned short f2bf(float f) {
  unsigned int u = __float_as_uint(f);
  u += 0x7FFFu + ((u >> 16) & 1u);          // round-to-nearest-even
  return (unsigned short)(u >> 16);
}

__device__ __forceinline__ unsigned int pack2bf(float a, float b) {
#if defined(__AMDGCN__) && __has_builtin(__builtin_amdgcn_cvt_pk_bf16_f32)
  v2bf p = __builtin_amdgcn_cvt_pk_bf16_f32(a, b);
  return __builtin_bit_cast(unsigned int, p);
#else
  return (unsigned)f2bf(a) | ((unsigned)f2bf(b) << 16);
#endif
}

__device__ __forceinline__ void wait_async0() {
#if defined(__AMDGCN__) && __has_builtin(__builtin_amdgcn_s_wait_asynccnt)
  __builtin_amdgcn_s_wait_asynccnt(0);
#elif defined(__AMDGCN__)
  asm volatile("s_wait_asynccnt 0" ::: "memory");
#endif
}

// build a 16-element bf16 fragment from two aligned 16B LDS reads
__device__ __forceinline__ v16bf ld_frag(const unsigned short* p0,
                                         const unsigned short* p1) {
  const uint4 q0 = *(const uint4*)p0;
  const uint4 q1 = *(const uint4*)p1;
  v8u t;
  t[0] = q0.x; t[1] = q0.y; t[2] = q0.z; t[3] = q0.w;
  t[4] = q1.x; t[5] = q1.y; t[6] = q1.z; t[7] = q1.w;
  return __builtin_bit_cast(v16bf, t);
}

// ---- f32 -> bf16 bulk conversion (packed) ----
__global__ __launch_bounds__(256)
void cvt_bf16_kernel(const float2* __restrict__ src, unsigned int* __restrict__ dst,
                     int n2) {
  const int i = blockIdx.x * 256 + threadIdx.x;
  if (i < n2) {
    const float2 v = src[i];
    dst[i] = pack2bf(v.x, v.y);
  }
}

// ---- tiled bf16 WMMA GEMM:  C[M,N] (op)= A[M,K] * W[N,K]^T (+bias) ----
// A, W pre-converted to bf16 (ushort). LDS staging, async copies if available.
#define BM   128
#define BN   64
#define BKT  32
#define LSTR 40          // BKT + 8 bf16 pad -> 80B row stride, 16B aligned frags

enum { EPI_NONE = 0, EPI_GELU = 1, EPI_ADD = 2 };

template <int MODE>
__global__ __launch_bounds__(256)
void gemm_bf16_wmma(const unsigned short* __restrict__ Ap,
                    const unsigned short* __restrict__ Wp,
                    const float* __restrict__ bias, float* __restrict__ Cp,
                    int N, int Kt) {
  __shared__ __align__(16) unsigned short As[BM * LSTR];
  __shared__ __align__(16) unsigned short Bs[BN * LSTR];

  const int tid  = threadIdx.x;
  const int lane = tid & 31;
  const int wv   = tid >> 5;     // 8 waves
  const int wm   = wv & 3;       // 4 M sub-tiles of 32
  const int wn   = wv >> 2;      // 2 N sub-tiles of 32
  const int lr   = lane & 15;
  const int hi   = lane >> 4;

  const int mBlock = blockIdx.x * BM;
  const int nBlock = blockIdx.y * BN;

  // per-thread tile-copy coordinates (16B = 8 bf16 chunks)
  const int rA = tid >> 2;              // rows 0..63 (second chunk: +64)
  const int cA = (tid & 3) << 3;        // bf16 col 0/8/16/24
  int nB = nBlock + rA;                 // weight row for B tile
  if (nB >= N) nB = N - 1;              // clamp; OOB cols guarded at store

  const size_t aBase0 = (size_t)(mBlock + rA) * Kt + cA;
  const size_t aBase1 = aBase0 + (size_t)64 * Kt;
  const size_t wBase  = (size_t)nB * Kt + cA;
  unsigned short* la0 = &As[rA * LSTR + cA];
  unsigned short* la1 = &As[(rA + 64) * LSTR + cA];
  unsigned short* lb  = &Bs[rA * LSTR + cA];

  v8f acc[2][2];
#pragma unroll
  for (int mi = 0; mi < 2; ++mi)
#pragma unroll
    for (int ni = 0; ni < 2; ++ni)
#pragma unroll
      for (int r = 0; r < 8; ++r) acc[mi][ni][r] = 0.0f;

  const int nk = Kt / BKT;
  for (int kb = 0; kb < nk; ++kb) {
    const int k0 = kb * BKT;
#if HAVE_ASYNC
    // CDNA5 async global->LDS copies, tracked by ASYNCcnt
    __builtin_amdgcn_global_load_async_to_lds_b128(AS1V(Ap + aBase0 + k0), AS3V(la0), 0, 0);
    __builtin_amdgcn_global_load_async_to_lds_b128(AS1V(Ap + aBase1 + k0), AS3V(la1), 0, 0);
    __builtin_amdgcn_global_load_async_to_lds_b128(AS1V(Wp + wBase  + k0), AS3V(lb),  0, 0);
    wait_async0();
#else
    *(uint4*)la0 = *(const uint4*)(Ap + aBase0 + k0);
    *(uint4*)la1 = *(const uint4*)(Ap + aBase1 + k0);
    uint4 wv4 = {0u, 0u, 0u, 0u};
    if (nBlock + rA < N) wv4 = *(const uint4*)(Wp + wBase + k0);
    *(uint4*)lb = wv4;
    if (kb + 1 < nk)
      __builtin_prefetch(Ap + aBase0 + k0 + BKT, 0, 1);
#endif
    __syncthreads();

    // fragments per ISA layouts: A lanes0-15 K{0..7,16..23}, lanes16-31 K{8..15,24..31}
    v16bf afr[2], bfr[2];
#pragma unroll
    for (int mi = 0; mi < 2; ++mi) {
      const unsigned short* p = &As[(wm * 32 + mi * 16 + lr) * LSTR + hi * 8];
      afr[mi] = ld_frag(p, p + 16);
    }
    // B: lane holds col n = lane%16, K {0..7,16..23} (lanes0-15) or {8..15,24..31} (lanes16-31)
#pragma unroll
    for (int ni = 0; ni < 2; ++ni) {
      const unsigned short* p = &Bs[(wn * 32 + ni * 16 + lr) * LSTR + hi * 8];
      bfr[ni] = ld_frag(p, p + 16);
    }
#pragma unroll
    for (int mi = 0; mi < 2; ++mi)
#pragma unroll
      for (int ni = 0; ni < 2; ++ni)
        acc[mi][ni] = __builtin_amdgcn_wmma_f32_16x16x32_bf16(
            false, afr[mi], false, bfr[ni], (short)0, acc[mi][ni], false, false);
    __syncthreads();
  }

  // epilogue: C VGPR r -> row (r + 8*hi), col lane%16
#pragma unroll
  for (int ni = 0; ni < 2; ++ni) {
    const int n = nBlock + wn * 32 + ni * 16 + lr;
    if (n >= N) continue;
    const float bv = bias ? bias[n] : 0.0f;
#pragma unroll
    for (int mi = 0; mi < 2; ++mi) {
      const int mbase = mBlock + wm * 32 + mi * 16 + hi * 8;
#pragma unroll
      for (int r = 0; r < 8; ++r) {
        float v = acc[mi][ni][r] + bv;
        if (MODE == EPI_GELU) v = 0.5f * v * (1.0f + erff(v * 0.70710678118654752f));
        const size_t off = (size_t)(mbase + r) * N + n;
        if (MODE == EPI_ADD) Cp[off] += v;
        else                 Cp[off] = v;
      }
    }
  }
}

// ---- embedding lookup + RoPE ----
__global__ __launch_bounds__(256)
void embed_rope_kernel(const int* __restrict__ tokens, const float* __restrict__ emb,
                       float* __restrict__ X) {
  const int row = blockIdx.x;            // b*L + l
  const int l   = row % L_;
  const float* e = emb + (size_t)tokens[row] * D_;
  for (int d = threadIdx.x; d < D_; d += 256) {
    const int dh = (d < HALF_) ? d : d - HALF_;
    const float inv  = __expf(-(float)dh * (9.210340371976184f / (float)HALF_));
    const float f    = (float)l * inv;
    float s, c;
    __sincosf(f, &s, &c);
    const float xv  = e[d];
    const float rot = (d < HALF_) ? -e[d + HALF_] : e[d - HALF_];
    X[(size_t)row * D_ + d] = xv * c + rot * s;
  }
}

// ---- RMSNorm (one row per block, 256 threads x 2 elems) ----
__global__ __launch_bounds__(256)
void rmsnorm_kernel(const float* __restrict__ X, const float* __restrict__ w,
                    float* __restrict__ Y) {
  const int row = blockIdx.x;
  const int t   = threadIdx.x;
  const float* x = X + (size_t)row * D_;
  const float v0 = x[t], v1 = x[t + 256];
  __shared__ float red[256];
  red[t] = v0 * v0 + v1 * v1;
  __syncthreads();
#pragma unroll
  for (int s = 128; s > 0; s >>= 1) {
    if (t < s) red[t] += red[t + s];
    __syncthreads();
  }
  const float inv = __frsqrt_rn(red[0] / (float)D_ + 1e-6f);
  Y[(size_t)row * D_ + t]       = w[t]       * v0 * inv;
  Y[(size_t)row * D_ + t + 256] = w[t + 256] * v1 * inv;
}

// ---- dilated depthwise conv (K=5, 'same'), writes (b,l,d) layout + bias ----
__global__ __launch_bounds__(256)
void dconv_kernel(const float* __restrict__ XN, const float* __restrict__ dw,
                  const float* __restrict__ db, float* __restrict__ XC, int dil) {
  const int idx = blockIdx.x * 256 + threadIdx.x;   // over B*L*D
  const int d = idx & (D_ - 1);
  const int l = (idx / D_) % L_;
  const int b = idx / (D_ * L_);
  const float* w = dw + d * K_;
  float acc = db[d];
#pragma unroll
  for (int k = 0; k < K_; ++k) {
    const int ls = l + (k - 2) * dil;
    if (0 <= ls && ls < L_)
      acc += w[k] * XN[((size_t)b * L_ + ls) * D_ + d];
  }
  XC[idx] = acc;
}

// ---- linear recurrence on first A channels; Y += hs (or += xn passthrough) ----
__global__ __launch_bounds__(512)
void recurrence_kernel(const float* __restrict__ XN, const float* __restrict__ al,
                       const float* __restrict__ be, float* __restrict__ Y) {
  const int b = blockIdx.x;
  const int d = threadIdx.x;     // 512 channels, coalesced across threads
  if (d < A_) {
    const float a = al[d], bb = be[d];
    float h = 0.f;
    for (int l = 0; l < L_; ++l) {
      const size_t off = ((size_t)b * L_ + l) * D_ + d;
      h = a * h + bb * XN[off];
      Y[off] += h;
    }
  } else {
    for (int l = 0; l < L_; ++l) {
      const size_t off = ((size_t)b * L_ + l) * D_ + d;
      Y[off] += XN[off];
    }
  }
}

// ---- gating: Y += sigmoid(G1) * tanh(G2) ----
__global__ __launch_bounds__(256)
void gate_kernel(float* __restrict__ Y, const float* __restrict__ G1,
                 const float* __restrict__ G2) {
  const int i = blockIdx.x * 256 + threadIdx.x;
  const float g1 = G1[i], g2 = G2[i];
  Y[i] += (1.0f / (1.0f + __expf(-g1))) * tanhf(g2);
}

// ---- residual add: X += Y ----
__global__ __launch_bounds__(256)
void resadd_kernel(float* __restrict__ X, const float* __restrict__ Y) {
  const int i = blockIdx.x * 256 + threadIdx.x;
  X[i] += Y[i];
}

extern "C" void kernel_launch(void* const* d_in, const int* in_sizes, int n_in,
                              void* d_out, int out_size, void* d_ws, size_t ws_size,
                              hipStream_t stream) {
  const int*   tokens  = (const int*)d_in[0];
  const float* emb     = (const float*)d_in[1];
  const float* norm_w  = (const float*)d_in[2];
  const float* dconv_w = (const float*)d_in[3];
  const float* dconv_b = (const float*)d_in[4];
  const float* pconv_w = (const float*)d_in[5];
  const float* pconv_b = (const float*)d_in[6];
  const float* alpha   = (const float*)d_in[7];
  const float* beta    = (const float*)d_in[8];
  const float* w1      = (const float*)d_in[9];
  const float* b1      = (const float*)d_in[10];
  const float* w2      = (const float*)d_in[11];
  const float* b2      = (const float*)d_in[12];
  const float* down_w  = (const float*)d_in[13];
  const float* down_b  = (const float*)d_in[14];
  const float* up_w    = (const float*)d_in[15];
  const float* up_b    = (const float*)d_in[16];
  const float* fnw     = (const float*)d_in[17];

  const size_t MD = (size_t)ML * D_;       // 2,097,152
  const size_t MH = (size_t)ML * H_;       //   524,288
  float* X  = (float*)d_ws;
  float* XN = X  + MD;
  float* XC = XN + MD;
  float* Y  = XC + MD;
  float* G1 = Y  + MD;
  float* G2 = G1 + MD;
  float* HD = G2 + MD;                     // ML x H
  unsigned short* Ab = (unsigned short*)(HD + MH);   // bf16 activations (<= MD)
  unsigned short* Wb = Ab + MD;                      // bf16 weights (<= D*D)
  unsigned short* Eb = Wb + (size_t)D_ * D_;         // bf16 embedding (V*D)

  auto cvt = [&](const float* s, unsigned short* d, size_t n) {
    const int n2 = (int)(n >> 1);
    cvt_bf16_kernel<<<(n2 + 255) / 256, 256, 0, stream>>>(
        (const float2*)s, (unsigned int*)d, n2);
  };

  const int nElem = ML * D_;
  const dim3 gD(ML / BM, D_ / BN);   // 32 x 8
  const dim3 gH(ML / BM, H_ / BN);   // 32 x 2

  embed_rope_kernel<<<ML, 256, 0, stream>>>(tokens, emb, X);

  for (int i = 0; i < NL_; ++i) {
    const int dil = 1 << (i % 3);
    rmsnorm_kernel<<<ML, 256, 0, stream>>>(X, norm_w + (size_t)i * D_, XN);
    dconv_kernel<<<nElem / 256, 256, 0, stream>>>(
        XN, dconv_w + (size_t)i * D_ * K_, dconv_b + (size_t)i * D_, XC, dil);

    cvt(XC, Ab, MD);
    cvt(pconv_w + (size_t)i * D_ * D_, Wb, (size_t)D_ * D_);
    gemm_bf16_wmma<EPI_NONE><<<gD, 256, 0, stream>>>(
        Ab, Wb, pconv_b + (size_t)i * D_, Y, D_, D_);

    recurrence_kernel<<<B_, D_, 0, stream>>>(
        XN, alpha + (size_t)i * A_, beta + (size_t)i * A_, Y);

    cvt(Y, Ab, MD);
    cvt(w1 + (size_t)i * D_ * D_, Wb, (size_t)D_ * D_);
    gemm_bf16_wmma<EPI_NONE><<<gD, 256, 0, stream>>>(
        Ab, Wb, b1 + (size_t)i * D_, G1, D_, D_);
    cvt(w2 + (size_t)i * D_ * D_, Wb, (size_t)D_ * D_);
    gemm_bf16_wmma<EPI_NONE><<<gD, 256, 0, stream>>>(
        Ab, Wb, b2 + (size_t)i * D_, G2, D_, D_);

    gate_kernel<<<nElem / 256, 256, 0, stream>>>(Y, G1, G2);

    cvt(Y, Ab, MD);
    cvt(down_w + (size_t)i * H_ * D_, Wb, (size_t)H_ * D_);
    gemm_bf16_wmma<EPI_GELU><<<gH, 256, 0, stream>>>(
        Ab, Wb, down_b + (size_t)i * H_, HD, H_, D_);

    cvt(HD, Ab, MH);
    cvt(up_w + (size_t)i * D_ * H_, Wb, (size_t)D_ * H_);
    gemm_bf16_wmma<EPI_ADD><<<gD, 256, 0, stream>>>(
        Ab, Wb, up_b + (size_t)i * D_, Y, D_, H_);

    resadd_kernel<<<nElem / 256, 256, 0, stream>>>(X, Y);
  }

  rmsnorm_kernel<<<ML, 256, 0, stream>>>(X, fnw, XN);
  cvt(XN, Ab, MD);
  cvt(emb, Eb, (size_t)V_ * D_);
  // logits: 4096 x 50257 x 512 — the dominant GEMM (~206 GFLOP), bf16 streams
  gemm_bf16_wmma<EPI_NONE><<<dim3(ML / BM, (V_ + BN - 1) / BN), 256, 0, stream>>>(
      Ab, Eb, nullptr, (float*)d_out, V_, D_);
}
